// HRNet_W48_Proto_Ultra_68994354643461
// MI455X (gfx1250) — matrix operand order repaired and checked by reading, not compiled, not run on hardware
//
#include <hip/hip_runtime.h>
#include <hip/hip_bf16.h>
#include <stdint.h>

// ---------------------------------------------------------------------------
// MI455X (gfx1250, wave32) implementation of the HRNet-W48 proto head.
// Heavy math (3x3 conv 720->720 @ 96x96x2, two 1x1 convs) runs as bf16 WMMA
// GEMMs (v_wmma_f32_16x16x32_bf16) with fp32 accumulate; BN folded into the
// GEMM epilogue. Channels padded 720->768 so K-steps of 32 never straddle a
// filter tap and N divides the 128-wide tile. Tiles are staged global->LDS
// with the CDNA5 async-DMA path (ASYNCcnt) when the builtins are available,
// double-buffered so the DMA of step k+1 overlaps the WMMAs of step k.
// ---------------------------------------------------------------------------

typedef __attribute__((ext_vector_type(16))) __bf16    v16bf;
typedef __attribute__((ext_vector_type(8)))  float     v8f;
typedef __attribute__((ext_vector_type(4)))  uint32_t  u32x4;
typedef __attribute__((__vector_size__(16))) int       v4i;

static constexpr int IMG   = 96;
static constexpr int HW    = IMG * IMG;        // 9216
static constexpr int NPIX  = 2 * HW;           // 18432
static constexpr int CIN   = 720;
static constexpr int CPAD  = 768;              // 24 * 32
static constexpr int K3    = 9 * CPAD;         // 6912
static constexpr int NPROT = 20;               // 2 classes * 10 protos

#if defined(__gfx1250__) && \
    __has_builtin(__builtin_amdgcn_global_load_async_to_lds_b128) && \
    __has_builtin(__builtin_amdgcn_s_wait_asynccnt)
#define USE_ASYNC_LDS 1
#endif

__device__ __forceinline__ uint16_t f2bf(float f) {
  uint32_t u = __builtin_bit_cast(uint32_t, f);
  uint32_t r = u + 0x7FFFu + ((u >> 16) & 1u);   // round-to-nearest-even
  return (uint16_t)(r >> 16);
}
__device__ __forceinline__ float bf2f(uint16_t h) {
  uint32_t u = ((uint32_t)h) << 16;
  return __builtin_bit_cast(float, u);
}
__device__ __forceinline__ float wred(float v) {
  #pragma unroll
  for (int o = 16; o > 0; o >>= 1) v += __shfl_xor(v, o, 32);
  return v;
}

// 16-byte global -> LDS copy: async DMA (ASYNCcnt) if available, else via VGPR
__device__ __forceinline__ void cp_b128(const uint16_t* g, uint16_t* l) {
#if defined(USE_ASYNC_LDS)
  __builtin_amdgcn_global_load_async_to_lds_b128(
      (__attribute__((address_space(1))) v4i*)g,
      (__attribute__((address_space(3))) v4i*)l, 0, 0);
#else
  *(u32x4*)l = *(const u32x4*)g;
#endif
}

// ---------------------------------------------------------------------------
// 1) Fused concat + bilinear(align_corners=True) upsample -> NHWC bf16, padded
// ---------------------------------------------------------------------------
__global__ __launch_bounds__(256) void k_build_feats(
    const float* __restrict__ f1, const float* __restrict__ f2,
    const float* __restrict__ f3, const float* __restrict__ f4,
    uint16_t* __restrict__ feats) {
  int idx = blockIdx.x * 256 + threadIdx.x;            // n * CPAD + c
  int n = idx / CPAD, c = idx - n * CPAD;
  int b = n / HW, yx = n - b * HW;
  int y = yx / IMG, x = yx - y * IMG;
  float v = 0.f;
  if (c < 48) {
    v = f1[((size_t)(b * 48 + c) * IMG + y) * IMG + x];
  } else if (c < CIN) {
    const float* src; int cc, S, Cs;
    if (c < 144)      { src = f2; cc = c - 48;  S = 48; Cs = 96;  }
    else if (c < 336) { src = f3; cc = c - 144; S = 24; Cs = 192; }
    else              { src = f4; cc = c - 336; S = 12; Cs = 384; }
    float sc = (float)(S - 1) / 95.0f;
    float ysf = y * sc, xsf = x * sc;
    int y0 = (int)ysf, x0 = (int)xsf;
    int y1 = min(y0 + 1, S - 1), x1 = min(x0 + 1, S - 1);
    float wy = ysf - (float)y0, wx = xsf - (float)x0;
    const float* p = src + ((size_t)b * Cs + cc) * S * S;
    float v00 = p[y0 * S + x0], v01 = p[y0 * S + x1];
    float v10 = p[y1 * S + x0], v11 = p[y1 * S + x1];
    v = (v00 * (1.f - wx) + v01 * wx) * (1.f - wy)
      + (v10 * (1.f - wx) + v11 * wx) * wy;
  }
  feats[idx] = f2bf(v);
}

// ---------------------------------------------------------------------------
// 2) Weight packing (bf16, co-major rows of K so B tiles load contiguously)
// ---------------------------------------------------------------------------
__global__ __launch_bounds__(256) void k_pack_w3(const float* __restrict__ w,
                                                 uint16_t* __restrict__ wt) {
  int idx = blockIdx.x * 256 + threadIdx.x;            // co * K3 + k
  int co = idx / K3, k = idx - co * K3;
  int tap = k / CPAD, ci = k - tap * CPAD;
  float v = (co < CIN && ci < CIN) ? w[((size_t)co * CIN + ci) * 9 + tap] : 0.f;
  wt[idx] = f2bf(v);
}

__global__ __launch_bounds__(256) void k_pack_w1(const float* __restrict__ w,
                                                 uint16_t* __restrict__ wt) {
  int idx = blockIdx.x * 256 + threadIdx.x;            // co * CPAD + ci
  int co = idx / CPAD, ci = idx - co * CPAD;
  float v = (co < CIN && ci < CIN) ? w[(size_t)co * CIN + ci] : 0.f;
  wt[idx] = f2bf(v);
}

// Per-channel alpha/beta (conv+BN fold) + zero pad block for OOB im2col rows
__global__ void k_pack_scales(
    const float* __restrict__ cls_b,
    const float* __restrict__ g1, const float* __restrict__ b1,
    const float* __restrict__ rm1, const float* __restrict__ rv1,
    const float* __restrict__ pb1,
    const float* __restrict__ g2, const float* __restrict__ b2,
    const float* __restrict__ rm2, const float* __restrict__ rv2,
    const float* __restrict__ pb2,
    float* __restrict__ ab, uint32_t* __restrict__ zpad) {
  int c = threadIdx.x;                                  // 0..767
  float a1 = 0.f, be1 = 0.f, a2 = 0.f, be2 = 0.f, a3 = 0.f, be3 = 0.f;
  if (c < CIN) {
    a1  = g1[c] * rsqrtf(rv1[c] + 1e-5f);
    be1 = (cls_b[c] - rm1[c]) * a1 + b1[c];
    a2  = g2[c] * rsqrtf(rv2[c] + 1e-5f);
    be2 = (pb1[c] - rm2[c]) * a2 + b2[c];
    a3  = 1.0f;
    be3 = pb2[c];
  }
  ab[c]            = a1;  ab[CPAD + c]     = be1;
  ab[2*CPAD + c]   = a2;  ab[3*CPAD + c]   = be2;
  ab[4*CPAD + c]   = a3;  ab[5*CPAD + c]   = be3;
  if (c < 64) zpad[c] = 0u;                             // 256 B of zeros
}

// ---------------------------------------------------------------------------
// 3) L2-normalize prototypes (20 rows of 720)
// ---------------------------------------------------------------------------
__global__ __launch_bounds__(256) void k_norm_protos(const float* __restrict__ proto,
                                                     float* __restrict__ out) {
  __shared__ float red[256];
  int p = blockIdx.x;
  const float* src = proto + (size_t)p * CIN;
  float s = 0.f;
  for (int c = threadIdx.x; c < CIN; c += 256) { float v = src[c]; s += v * v; }
  red[threadIdx.x] = s; __syncthreads();
  #pragma unroll
  for (int o = 128; o > 0; o >>= 1) {
    if ((int)threadIdx.x < o) red[threadIdx.x] += red[threadIdx.x + o];
    __syncthreads();
  }
  float inv = 1.0f / fmaxf(sqrtf(red[0]), 1e-12f);
  for (int c = threadIdx.x; c < CIN; c += 256) out[(size_t)p * CIN + c] = src[c] * inv;
}

// ---------------------------------------------------------------------------
// 4) WMMA GEMM: Out[M x CPAD] = act(A(im2col) x W^T * alpha + beta)
//    Block tile 128x128x32, 8 waves (4Mx2N), wave tile 32x64 = 2x4 frags.
//    Double-buffered LDS staged via async global->LDS DMA.
// ---------------------------------------------------------------------------
template <int KSIZE, bool CONV3, bool RELU>
__global__ __launch_bounds__(256) void k_gemm(
    const uint16_t* __restrict__ A, const uint16_t* __restrict__ W,
    const float* __restrict__ alpha, const float* __restrict__ beta,
    const uint16_t* __restrict__ zpad,
    uint16_t* __restrict__ Out) {
  // 80 B LDS row stride (32 bf16 + 8 pad): keeps b128 alignment, breaks the
  // 64-bank power-of-two conflict pattern. Two buffers for DMA/WMMA overlap.
  __shared__ __align__(16) uint16_t sA[2][128 * 40];
  __shared__ __align__(16) uint16_t sB[2][128 * 40];

  const int tid   = threadIdx.x;
  const int lane  = tid & 31;
  const int wid   = tid >> 5;
  const int waveM = wid & 3;        // 4 groups of 32 rows
  const int waveN = wid >> 2;       // 2 groups of 64 cols
  const int mbase = blockIdx.x * 128;
  const int nbase = blockIdx.y * 128;

  // tile-load mapping: each thread moves 32 B (16 bf16) of one row
  const int lr = tid >> 1;          // 0..127
  const int lk = (tid & 1) * 16;    // element offset within 32-wide K slab

  // pixel decomposition of this thread's A row (for 3x3 im2col)
  const int an  = mbase + lr;
  const int abt = an / HW;
  const int ayx = an - abt * HW;
  const int ay  = ayx / IMG, ax = ayx - ay * IMG;

  const uint16_t* brow = W + (size_t)(nbase + lr) * KSIZE + lk;

  // A row base pointer for a given tap (recomputed once per 24 K-steps);
  // out-of-image rows read from a 256 B zero block instead.
  auto rowptr = [&](int tap) -> const uint16_t* {
    if (!CONV3) return A + (size_t)an * CPAD + lk;
    int ys = ay + tap / 3 - 1;
    int xs = ax + tap % 3 - 1;
    bool ok = (ys >= 0) && (ys < IMG) && (xs >= 0) && (xs < IMG);
    return ok ? (A + ((size_t)(abt * HW + ys * IMG + xs)) * CPAD + lk)
              : (zpad + lk);
  };

  int s_k = 0, s_ci = 0, s_tap = 0;      // staging cursor
  const uint16_t* s_arow = rowptr(0);

  auto stage = [&](int buf) {
    uint16_t* la = &sA[buf][lr * 40 + lk];
    uint16_t* lb = &sB[buf][lr * 40 + lk];
    const uint16_t* ga = s_arow + s_ci;
    const uint16_t* gb = brow + s_k;
    cp_b128(ga, la);     cp_b128(ga + 8, la + 8);
    cp_b128(gb, lb);     cp_b128(gb + 8, lb + 8);
  };
  auto advance = [&]() {
    s_k += 32; s_ci += 32;
    if (CONV3 && s_ci == CPAD) {
      s_ci = 0; ++s_tap;
      if (s_tap < 9) s_arow = rowptr(s_tap);
    }
  };

  v8f acc[2][4];
  const v8f vz = {0.f, 0.f, 0.f, 0.f, 0.f, 0.f, 0.f, 0.f};
  #pragma unroll
  for (int i = 0; i < 2; i++)
    #pragma unroll
    for (int j = 0; j < 4; j++) acc[i][j] = vz;

  const int lrow  = lane & 15;
  const int lhalf = lane >> 4;

  constexpr int NS = KSIZE / 32;
  stage(0); advance();

  for (int ks = 0; ks < NS; ks++) {
    const int cur = ks & 1;
    const bool more = (ks + 1 < NS);
    if (more) { stage(cur ^ 1); advance(); }   // DMA next slab while computing
#if defined(USE_ASYNC_LDS)
    if (more) __builtin_amdgcn_s_wait_asynccnt(4);  // our 4 oldest DMAs landed
    else      __builtin_amdgcn_s_wait_asynccnt(0);
#endif
    __syncthreads();   // all waves' slabs for `cur` visible

    // ---- LDS -> fragments (per ISA lane layouts) ----
    union FU { u32x4 q[2]; v16bf v; } fa[2], fb[4];
    #pragma unroll
    for (int i = 0; i < 2; i++) {
      // A 16x32: lanes 0-15 hold K {0..7,16..23}; lanes 16-31 {8..15,24..31}
      const uint16_t* ap = &sA[cur][(waveM * 32 + i * 16 + lrow) * 40 + lhalf * 8];
      fa[i].q[0] = *(const u32x4*)ap;
      fa[i].q[1] = *(const u32x4*)(ap + 16);
    }
    #pragma unroll
    for (int j = 0; j < 4; j++) {
      // B 32x16: lanes 0-15 hold K {0..7,16..23}; lanes 16-31 {8..15,24..31}
      const uint16_t* bp = &sB[cur][(waveN * 64 + j * 16 + lrow) * 40 + lhalf * 8];
      fb[j].q[0] = *(const u32x4*)bp;
      fb[j].q[1] = *(const u32x4*)(bp + 16);
    }
    #pragma unroll
    for (int i = 0; i < 2; i++)
      #pragma unroll
      for (int j = 0; j < 4; j++)
        acc[i][j] = __builtin_amdgcn_wmma_f32_16x16x32_bf16(
            false, fa[i].v, false, fb[j].v, (short)0, acc[i][j], false, false);

    __syncthreads();   // done reading `cur` before it is re-staged
  }

  // ---- epilogue: y = acc*alpha[n] + beta[n], optional ReLU, bf16 store ----
  const int mrow0 = mbase + waveM * 32 + (lhalf << 3);
  #pragma unroll
  for (int j = 0; j < 4; j++) {
    int nc = nbase + waveN * 64 + j * 16 + lrow;
    float al = alpha[nc], be = beta[nc];
    #pragma unroll
    for (int i = 0; i < 2; i++) {
      int mr = mrow0 + i * 16;
      #pragma unroll
      for (int r = 0; r < 8; r++) {
        float v = acc[i][j][r] * al + be;
        if (RELU) v = fmaxf(v, 0.f);
        Out[(size_t)(mr + r) * CPAD + nc] = f2bf(v);
      }
    }
  }
}

// ---------------------------------------------------------------------------
// 5) Per-pixel head: LN(720) -> L2 -> proto cosine -> max -> LN(2)
//    One wave per pixel; prototypes + LN params staged in LDS per block.
// ---------------------------------------------------------------------------
__global__ __launch_bounds__(256) void k_head(
    const uint16_t* __restrict__ P2, const float* __restrict__ protoN,
    const float* __restrict__ lg, const float* __restrict__ lb,
    const float* __restrict__ mg, const float* __restrict__ mb,
    float* __restrict__ out) {
  __shared__ float sp[NPROT * CIN];
  __shared__ float slg[CIN], slb[CIN];
  for (int i = threadIdx.x; i < NPROT * CIN; i += 256) sp[i] = protoN[i];
  for (int i = threadIdx.x; i < CIN; i += 256) { slg[i] = lg[i]; slb[i] = lb[i]; }
  __syncthreads();

  const int lane = threadIdx.x & 31;
  const int wid  = threadIdx.x >> 5;
  const int n    = blockIdx.x * 8 + wid;

  float xv[23];
  float s = 0.f, s2 = 0.f;
  #pragma unroll
  for (int i = 0; i < 23; i++) {
    int c = lane + 32 * i;
    float v = (c < CIN) ? bf2f(P2[(size_t)n * CPAD + c]) : 0.f;
    xv[i] = v; s += v; s2 += v * v;
  }
  s = wred(s); s2 = wred(s2);
  const float mu   = s * (1.0f / CIN);
  const float var  = s2 * (1.0f / CIN) - mu * mu;
  const float rstd = rsqrtf(var + 1e-5f);

  float l2 = 0.f;
  #pragma unroll
  for (int i = 0; i < 23; i++) {
    int c = lane + 32 * i;
    float xn = (c < CIN) ? (xv[i] - mu) * rstd * slg[c] + slb[c] : 0.f;
    xv[i] = xn; l2 += xn * xn;
  }
  l2 = wred(l2);
  const float rinv = 1.0f / fmaxf(sqrtf(l2), 1e-12f);

  float m0 = -1e30f, m1 = -1e30f;
  for (int p = 0; p < NPROT; p++) {
    const float* pp = &sp[p * CIN];
    float d = 0.f;
    #pragma unroll
    for (int i = 0; i < 23; i++) {
      int c = lane + 32 * i;
      if (c < CIN) d += xv[i] * pp[c];
    }
    d = wred(d) * rinv;
    if (p < 10) m0 = fmaxf(m0, d); else m1 = fmaxf(m1, d);
  }
  if (lane == 0) {
    float mean = 0.5f * (m0 + m1);
    float v0 = m0 - mean, v1 = m1 - mean;
    float rs = rsqrtf(0.5f * (v0 * v0 + v1 * v1) + 1e-5f);
    int b = n / HW, yx = n - b * HW;
    out[(size_t)(b * 2 + 0) * HW + yx] = v0 * rs * mg[0] + mb[0];
    out[(size_t)(b * 2 + 1) * HW + yx] = v1 * rs * mg[1] + mb[1];
  }
}

// ---------------------------------------------------------------------------
// Launch
// ---------------------------------------------------------------------------
extern "C" void kernel_launch(void* const* d_in, const int* in_sizes, int n_in,
                              void* d_out, int out_size, void* d_ws, size_t ws_size,
                              hipStream_t stream) {
  (void)in_sizes; (void)n_in; (void)out_size; (void)ws_size;
  const float* f1      = (const float*)d_in[0];
  const float* f2      = (const float*)d_in[1];
  const float* f3      = (const float*)d_in[2];
  const float* f4      = (const float*)d_in[3];
  const float* cls_w   = (const float*)d_in[4];
  const float* cls_b   = (const float*)d_in[5];
  const float* cbn_g   = (const float*)d_in[6];
  const float* cbn_b   = (const float*)d_in[7];
  const float* cbn_rm  = (const float*)d_in[8];
  const float* cbn_rv  = (const float*)d_in[9];
  const float* pw1     = (const float*)d_in[10];
  const float* pb1     = (const float*)d_in[11];
  const float* pbn_g   = (const float*)d_in[12];
  const float* pbn_b   = (const float*)d_in[13];
  const float* pbn_rm  = (const float*)d_in[14];
  const float* pbn_rv  = (const float*)d_in[15];
  const float* pw2     = (const float*)d_in[16];
  const float* pb2     = (const float*)d_in[17];
  const float* lnf_g   = (const float*)d_in[18];
  const float* lnf_b   = (const float*)d_in[19];
  const float* lnm_g   = (const float*)d_in[20];
  const float* lnm_b   = (const float*)d_in[21];
  const float* protos  = (const float*)d_in[22];

  // workspace carve-up (all 256-aligned); feats reused as p1, cbuf as p2
  char* ws = (char*)d_ws;
  const size_t SZ_FEATS = (size_t)NPIX * CPAD * 2;   // 28,311,552
  const size_t SZ_WT3   = (size_t)CPAD * K3 * 2;     // 10,616,832
  const size_t SZ_WT1   = (size_t)CPAD * CPAD * 2;   //  1,179,648
  uint16_t* feats  = (uint16_t*)(ws);
  uint16_t* wt3    = (uint16_t*)(ws + SZ_FEATS);
  uint16_t* cbuf   = (uint16_t*)(ws + SZ_FEATS + SZ_WT3);
  uint16_t* wt1    = (uint16_t*)(ws + 2 * SZ_FEATS + SZ_WT3);
  uint16_t* wt2    = (uint16_t*)(ws + 2 * SZ_FEATS + SZ_WT3 + SZ_WT1);
  float*    ab     = (float*)   (ws + 2 * SZ_FEATS + SZ_WT3 + 2 * SZ_WT1);
  float*    protoN = (float*)   (ws + 2 * SZ_FEATS + SZ_WT3 + 2 * SZ_WT1
                                    + (size_t)6 * CPAD * 4);
  uint32_t* zpad   = (uint32_t*)(ws + 2 * SZ_FEATS + SZ_WT3 + 2 * SZ_WT1
                                    + (size_t)6 * CPAD * 4
                                    + (size_t)NPROT * CIN * 4);

  k_build_feats<<<(NPIX * CPAD) / 256, 256, 0, stream>>>(f1, f2, f3, f4, feats);
  k_pack_w3<<<(CPAD * K3) / 256, 256, 0, stream>>>(cls_w, wt3);
  k_pack_w1<<<(CPAD * CPAD) / 256, 256, 0, stream>>>(pw1, wt1);
  k_pack_w1<<<(CPAD * CPAD) / 256, 256, 0, stream>>>(pw2, wt2);
  k_pack_scales<<<1, CPAD, 0, stream>>>(cls_b, cbn_g, cbn_b, cbn_rm, cbn_rv,
                                        pb1, pbn_g, pbn_b, pbn_rm, pbn_rv,
                                        pb2, ab, zpad);
  k_norm_protos<<<NPROT, 256, 0, stream>>>(protos, protoN);

  dim3 gg(NPIX / 128, CPAD / 128);   // 144 x 6
  // c = relu(bn(conv3x3(feats)))
  k_gemm<K3, true, true><<<gg, 256, 0, stream>>>(
      feats, wt3, ab, ab + CPAD, (const uint16_t*)zpad, cbuf);
  // p1 = relu(bn(conv1x1(c)))  -> reuse feats buffer
  k_gemm<CPAD, false, true><<<gg, 256, 0, stream>>>(
      cbuf, wt1, ab + 2 * CPAD, ab + 3 * CPAD, (const uint16_t*)zpad, feats);
  // p2 = conv1x1(p1) + b       -> reuse cbuf
  k_gemm<CPAD, false, false><<<gg, 256, 0, stream>>>(
      feats, wt2, ab + 4 * CPAD, ab + 5 * CPAD, (const uint16_t*)zpad, cbuf);

  k_head<<<NPIX / 8, 256, 0, stream>>>(cbuf, protoN, lnf_g, lnf_b,
                                       lnm_g, lnm_b, (float*)d_out);
}
